// CrossformerKeras_55568286875915
// MI455X (gfx1250) — matrix operand order repaired and checked by reading, not compiled, not run on hardware
//
#include <hip/hip_runtime.h>

// ---------------------------------------------------------------------------
// Crossformer forward for MI455X (gfx1250, wave32).
// All heavy GEMMs -> v_wmma_f32_16x16x32_f16 (f32->f16 convert on LDS stage,
// f32 accumulate). Attention core / LN / permutes are VALU kernels.
// ---------------------------------------------------------------------------

#define B_      64
#define DD      32     // data_dim
#define DM      512    // d_model
#define DFF     1024
#define NH      8
#define HD      64
#define FACTOR_ 10
#define ISEG    48
#define OSEG    24
#define SL      24     // seg_len

typedef __attribute__((ext_vector_type(8)))  _Float16 v8h;
typedef __attribute__((ext_vector_type(16))) _Float16 v16h;
typedef __attribute__((ext_vector_type(8)))  float    v8f;

// =============================== GEMM (WMMA) ===============================
#define BM  128
#define BN  128
#define BK  32
#define LDK 40   // BK + 8 halfs pad -> 80B row stride (16B aligned, bank-safe)

__device__ __forceinline__ v16h ld_frag(const _Float16* p0, const _Float16* p1) {
  v8h lo = *(const v8h*)p0;
  v8h hi = *(const v8h*)p1;
  v16h r;
#pragma unroll
  for (int i = 0; i < 8; ++i) { r[i] = lo[i]; r[i + 8] = hi[i]; }
  return r;
}

// C[M,N] = act(A[M,K] @ B[K,N] + bias)   flags: bit0 = GELU, bit1 = C += (accumulate)
__global__ __launch_bounds__(256) void gemm_wmma_kernel(
    float* __restrict__ C, const float* __restrict__ A, const float* __restrict__ Bm,
    const float* __restrict__ bias, int M, int N, int K, int flags) {
  __shared__ _Float16 sA[BM][LDK];
  __shared__ _Float16 sB[BN][LDK];   // B stored transposed: sB[n][k]
  const int tid  = threadIdx.x;
  const int lane = tid & 31;
  const int wave = tid >> 5;
  const int wm   = wave & 3;   // 4 waves along M (32 rows each)
  const int wn   = wave >> 2;  // 2 waves along N (64 cols each)
  const int bm0  = blockIdx.y * BM;
  const int bn0  = blockIdx.x * BN;

  v8f acc[2][4];
#pragma unroll
  for (int i = 0; i < 2; ++i)
#pragma unroll
    for (int j = 0; j < 4; ++j) acc[i][j] = v8f{};

  const int arow  = lane & 15;
  const int khalf = (lane >> 4) << 3;   // A frag: lanes 16-31 start at K=8
  const int koff  = (lane >> 4) << 4;   // (unused) B frag uses khalf, same as A

  for (int k0 = 0; k0 < K; k0 += BK) {
    // prefetch next K-tile (CDNA5 global_prefetch_b8)
    if (k0 + BK < K) {
      int m = tid >> 5, kk = tid & 31;
      int gm = bm0 + m, gk = k0 + BK + kk;
      if (gm < M && gk < K) __builtin_prefetch(&A[(size_t)gm * K + gk], 0, 0);
      int nn = tid & 127, kb = tid >> 7;
      int gn = bn0 + nn, gkb = k0 + BK + kb;
      if (gn < N && gkb < K) __builtin_prefetch(&Bm[(size_t)gkb * N + gn], 0, 0);
    }
    // stage A tile (f32 -> f16)
#pragma unroll
    for (int i = 0; i < 16; ++i) {
      int idx = (i << 8) + tid;
      int m = idx >> 5, kk = idx & 31;
      int gm = bm0 + m, gk = k0 + kk;
      float v = (gm < M && gk < K) ? A[(size_t)gm * K + gk] : 0.0f;
      sA[m][kk] = (_Float16)v;
    }
    // stage B tile transposed (coalesced along N)
#pragma unroll
    for (int i = 0; i < 16; ++i) {
      int idx = (i << 8) + tid;
      int n = idx & 127, kk = idx >> 7;
      int gn = bn0 + n, gk = k0 + kk;
      float v = (gn < N && gk < K) ? Bm[(size_t)gk * N + gn] : 0.0f;
      sB[n][kk] = (_Float16)v;
    }
    __syncthreads();

    v16h bf[4];
#pragma unroll
    for (int j = 0; j < 4; ++j) {
      const _Float16* pb = &sB[(wn << 6) + (j << 4) + arow][khalf];
      bf[j] = ld_frag(pb, pb + 16);
    }
    v16h af[2];
#pragma unroll
    for (int i = 0; i < 2; ++i) {
      const _Float16* pa = &sA[(wm << 5) + (i << 4) + arow][khalf];
      af[i] = ld_frag(pa, pa + 16);
    }
#pragma unroll
    for (int i = 0; i < 2; ++i)
#pragma unroll
      for (int j = 0; j < 4; ++j)
        acc[i][j] = __builtin_amdgcn_wmma_f32_16x16x32_f16(
            false, af[i], false, bf[j], (short)0, acc[i][j], false, false);
    __syncthreads();
  }

  // epilogue: C layout -> VGPR r: rows r (lanes 0-15) / 8+r (lanes 16-31), col = lane&15
  const int rhi = (lane >> 4) << 3;
  const int cl  = lane & 15;
#pragma unroll
  for (int i = 0; i < 2; ++i) {
#pragma unroll
    for (int j = 0; j < 4; ++j) {
      int col = bn0 + (wn << 6) + (j << 4) + cl;
      if (col >= N) continue;
      float bv = bias ? bias[col] : 0.0f;
      int rbase = bm0 + (wm << 5) + (i << 4) + rhi;
#pragma unroll
      for (int r = 0; r < 8; ++r) {
        int gm = rbase + r;
        if (gm < M) {
          float v = acc[i][j][r] + bv;
          if (flags & 1) v = 0.5f * v * (1.0f + erff(v * 0.70710678118654752f));
          size_t o = (size_t)gm * N + col;
          if (flags & 2) C[o] += v; else C[o] = v;
        }
      }
    }
  }
}

// =============================== LayerNorm =================================
// One wave per row; dim = PER*32. Optional fused residual: out = LN(x + res).
template <int PER>
__global__ __launch_bounds__(256) void ln_kernel(
    float* __restrict__ out, const float* __restrict__ x, const float* __restrict__ res,
    const float* __restrict__ g, const float* __restrict__ b, int rows, float eps) {
  const int dim = PER * 32;
  int row  = blockIdx.x * 8 + (threadIdx.x >> 5);
  int lane = threadIdx.x & 31;
  if (row >= rows) return;
  const float* px = x + (size_t)row * dim;
  const float* pr = res ? res + (size_t)row * dim : nullptr;
  float vals[PER];
  float s = 0.0f;
#pragma unroll
  for (int i = 0; i < PER; ++i) {
    float v = px[lane + 32 * i];
    if (pr) v += pr[lane + 32 * i];
    vals[i] = v; s += v;
  }
#pragma unroll
  for (int o = 16; o > 0; o >>= 1) s += __shfl_xor(s, o, 32);
  float mu = s / dim;
  float vs = 0.0f;
#pragma unroll
  for (int i = 0; i < PER; ++i) { float d = vals[i] - mu; vs += d * d; }
#pragma unroll
  for (int o = 16; o > 0; o >>= 1) vs += __shfl_xor(vs, o, 32);
  float rstd = rsqrtf(vs / dim + eps);
  float* po = out + (size_t)row * dim;
#pragma unroll
  for (int i = 0; i < PER; ++i) {
    int c = lane + 32 * i;
    po[c] = (vals[i] - mu) * rstd * g[c] + b[c];
  }
}

// ========================= Attention core (per g,h) ========================
#define MAXL 48
__global__ __launch_bounds__(64) void attn_kernel(
    float* __restrict__ O, const float* __restrict__ Q, const float* __restrict__ K,
    const float* __restrict__ V, int Lq, int Lk, float scale) {
  __shared__ float sQ[MAXL][HD], sK[MAXL][HD], sV[MAXL][HD], sS[MAXL][MAXL];
  int g = blockIdx.x >> 3, h = blockIdx.x & 7;
  int tid = threadIdx.x;
  for (int i = tid; i < Lq * HD; i += 64) {
    int l = i >> 6, c = i & 63;
    sQ[l][c] = Q[(size_t)(g * Lq + l) * DM + h * HD + c];
  }
  for (int i = tid; i < Lk * HD; i += 64) {
    int l = i >> 6, c = i & 63;
    size_t o = (size_t)(g * Lk + l) * DM + h * HD + c;
    sK[l][c] = K[o]; sV[l][c] = V[o];
  }
  __syncthreads();
  for (int i = tid; i < Lq * Lk; i += 64) {
    int r = i / Lk, cc = i - r * Lk;
    float s = 0.0f;
#pragma unroll
    for (int c = 0; c < HD; ++c) s += sQ[r][c] * sK[cc][c];
    sS[r][cc] = s * scale;
  }
  __syncthreads();
  for (int r = tid; r < Lq; r += 64) {
    float m = -3.0e38f;
    for (int j = 0; j < Lk; ++j) m = fmaxf(m, sS[r][j]);
    float s = 0.0f;
    for (int j = 0; j < Lk; ++j) { float e = expf(sS[r][j] - m); sS[r][j] = e; s += e; }
    float inv = 1.0f / s;
    for (int j = 0; j < Lk; ++j) sS[r][j] *= inv;
  }
  __syncthreads();
  for (int i = tid; i < Lq * HD; i += 64) {
    int r = i >> 6, c = i & 63;
    float o = 0.0f;
    for (int j = 0; j < Lk; ++j) o += sS[r][j] * sV[j][c];
    O[(size_t)(g * Lq + r) * DM + h * HD + c] = o;
  }
}

// =========================== utility kernels ===============================
__global__ void permute4_kernel(float* __restrict__ dst, const float* __restrict__ src,
                                long long n, int d1, int d2, int d3,
                                long long s0, long long s1, long long s2, long long s3) {
  long long i = (long long)blockIdx.x * blockDim.x + threadIdx.x;
  if (i >= n) return;
  int i3 = (int)(i % d3); long long t = i / d3;
  int i2 = (int)(t % d2); t /= d2;
  int i1 = (int)(t % d1); long long i0 = t / d1;
  dst[i] = src[i0 * s0 + i1 * s1 + i2 * s2 + i3 * s3];
}

__global__ void bcast_kernel(float* __restrict__ dst, const float* __restrict__ src,
                             long long n, long long chunk, int add) {
  long long i = (long long)blockIdx.x * blockDim.x + threadIdx.x;
  if (i >= n) return;
  float v = src[i % chunk];
  dst[i] = add ? dst[i] + v : v;
}

__global__ void fill0_kernel(float* __restrict__ dst, long long n) {
  long long i = (long long)blockIdx.x * blockDim.x + threadIdx.x;
  if (i < n) dst[i] = 0.0f;
}

// ============================ host-side params =============================
struct Dense { const float* b; const float* w; };
struct LNP   { const float* b; const float* g; };
struct AttnP { Dense k, o, q, v; };
struct TSAP  {
  Dense m1l1, m1l2, m2l1, m2l2;
  LNP n1, n2, n3, n4;
  AttnP recv; const float* router; AttnP send, time;
};
struct DecP { AttnP cross; Dense ml1, ml2; LNP n1, n2; Dense pred; TSAP tsa; };

static Dense getDense(void* const* din, int i) { return { (const float*)din[i], (const float*)din[i + 1] }; }
static LNP   getLN(void* const* din, int i)    { return { (const float*)din[i], (const float*)din[i + 1] }; }
static AttnP getAttn(void* const* din, int i) {
  AttnP a; a.k = getDense(din, i); a.o = getDense(din, i + 2);
  a.q = getDense(din, i + 4); a.v = getDense(din, i + 6); return a;
}
static TSAP getTSA(void* const* din, int t) {
  TSAP p;
  p.m1l1 = getDense(din, t + 0);  p.m1l2 = getDense(din, t + 2);
  p.m2l1 = getDense(din, t + 4);  p.m2l2 = getDense(din, t + 6);
  p.n1 = getLN(din, t + 8);  p.n2 = getLN(din, t + 10);
  p.n3 = getLN(din, t + 12); p.n4 = getLN(din, t + 14);
  p.recv = getAttn(din, t + 16);
  p.router = (const float*)din[t + 24];
  p.send = getAttn(din, t + 25);
  p.time = getAttn(din, t + 33);
  return p;
}
static DecP getDec(void* const* din, int l) {
  DecP d;
  d.cross = getAttn(din, l + 0);
  d.ml1 = getDense(din, l + 8);  d.ml2 = getDense(din, l + 10);
  d.n1 = getLN(din, l + 12);     d.n2 = getLN(din, l + 14);
  d.pred = getDense(din, l + 16);
  d.tsa = getTSA(din, l + 18);
  return d;
}

// ============================ launch wrappers ==============================
static inline void gemm(hipStream_t st, float* C, const float* A, const float* B,
                        const float* bias, int M, int N, int K, int flags) {
  dim3 grid((N + BN - 1) / BN, (M + BM - 1) / BM);
  gemm_wmma_kernel<<<grid, dim3(256), 0, st>>>(C, A, B, bias, M, N, K, flags);
}
static inline void ln512(hipStream_t st, float* out, const float* x, const float* res,
                         const LNP& p, int rows, float eps = 1e-5f) {
  ln_kernel<16><<<(rows + 7) / 8, 256, 0, st>>>(out, x, res, p.g, p.b, rows, eps);
}
static inline void ln2048(hipStream_t st, float* out, const float* x, const float* res,
                          const LNP& p, int rows, float eps = 1e-5f) {
  ln_kernel<64><<<(rows + 7) / 8, 256, 0, st>>>(out, x, res, p.g, p.b, rows, eps);
}
static inline void attention(hipStream_t st, float* O, const float* Q, const float* K,
                             const float* V, int G, int Lq, int Lk) {
  attn_kernel<<<G * NH, 64, 0, st>>>(O, Q, K, V, Lq, Lk, 0.125f);  // 1/sqrt(64)
}
static inline void permute4(hipStream_t st, float* dst, const float* src, long long n,
                            int d1, int d2, int d3,
                            long long s0, long long s1, long long s2, long long s3) {
  permute4_kernel<<<(unsigned)((n + 255) / 256), 256, 0, st>>>(dst, src, n, d1, d2, d3, s0, s1, s2, s3);
}
static inline void bcast(hipStream_t st, float* dst, const float* src, long long n,
                         long long chunk, int add) {
  bcast_kernel<<<(unsigned)((n + 255) / 256), 256, 0, st>>>(dst, src, n, chunk, add);
}
static inline void fill0(hipStream_t st, float* dst, long long n) {
  fill0_kernel<<<(unsigned)((n + 255) / 256), 256, 0, st>>>(dst, n);
}

struct Work { float *S0, *S1, *S2, *S3, *H, *RB, *RQ, *RA, *RO; };

static void mha(hipStream_t st, float* out, const float* qin, const float* kvin,
                int G, int Lq, int Lk, const AttnP& p,
                float* Qb, float* Kb, float* Vb, float* Ab) {
  gemm(st, Qb, qin,  p.q.w, p.q.b, G * Lq, DM, DM, 0);
  gemm(st, Kb, kvin, p.k.w, p.k.b, G * Lk, DM, DM, 0);
  gemm(st, Vb, kvin, p.v.w, p.v.b, G * Lk, DM, DM, 0);
  attention(st, Ab, Qb, Kb, Vb, G, Lq, Lk);
  gemm(st, out, Ab, p.o.w, p.o.b, G * Lq, DM, DM, 0);  // out may alias Qb (stream-ordered)
}

// Two-stage attention block. in/out: (b, d, seg, DM), both preserved vs scratch.
static void tsa_fwd(hipStream_t st, float* out, const float* in, int b, int d, int seg,
                    const TSAP& p, const Work& w) {
  const int rows = b * d * seg;
  // stage 1: time attention over each series
  mha(st, w.S0, in, in, b * d, seg, seg, p.time, w.S0, w.S1, w.S2, w.S3);
  ln512(st, w.S3, in, w.S0, p.n1, rows);                              // t
  gemm(st, w.H,  w.S3, p.m1l1.w, p.m1l1.b, rows, DFF, DM, 1);        // GELU
  gemm(st, w.S0, w.H,  p.m1l2.w, p.m1l2.b, rows, DM, DFF, 0);
  ln512(st, w.S1, w.S3, w.S0, p.n2, rows);                           // t2
  // stage 2: (b,d,seg,dm) -> (b,seg,d,dm)
  permute4(st, w.S2, w.S1, (long long)rows * DM, seg, d, DM,
           (long long)d * seg * DM, DM, (long long)seg * DM, 1);
  bcast(st, w.RB, p.router, (long long)b * seg * FACTOR_ * DM,
        (long long)seg * FACTOR_ * DM, 0);
  // send: router gathers from series
  mha(st, w.RO, w.RB, w.S2, b * seg, FACTOR_, d, p.send, w.RQ, w.S0, w.S3, w.RA);
  // recv: series read back from router buffer
  mha(st, w.S0, w.S2, w.RO, b * seg, d, FACTOR_, p.recv, w.S0, w.RQ, w.RB, w.S3);
  ln512(st, w.S3, w.S2, w.S0, p.n3, rows);                           // s1
  gemm(st, w.H,  w.S3, p.m2l1.w, p.m2l1.b, rows, DFF, DM, 1);
  gemm(st, w.S0, w.H,  p.m2l2.w, p.m2l2.b, rows, DM, DFF, 0);
  ln512(st, w.S1, w.S3, w.S0, p.n4, rows);                           // s2
  // back: (b,seg,d,dm) -> (b,d,seg,dm)
  permute4(st, out, w.S1, (long long)rows * DM, d, seg, DM,
           (long long)seg * d * DM, DM, (long long)d * DM, 1);
}

// ================================ forward ==================================
// JAX pytree flatten order (dict keys sorted alphabetically):
//   dec_layers[0..3] (59 each) | dec_pos=236 | dsw.b=237 dsw.w=238 |
//   enc0.tsa=239 | enc1: lin.b=280 lin.w=281 norm.b=282 norm.g=283 tsa=284 |
//   enc2: lin.b=325 lin.w=326 norm.b=327 norm.g=328 tsa=329 |
//   enc_pos=370 | pre_norm.b=371 pre_norm.g=372 | x_seq=373
extern "C" void kernel_launch(void* const* d_in, const int* in_sizes, int n_in,
                              void* d_out, int out_size, void* d_ws, size_t ws_size,
                              hipStream_t stream) {
  (void)in_sizes; (void)out_size; (void)ws_size;
  if (n_in < 374) return;
  hipStream_t st = stream;

  float* ws = (float*)d_ws;
  size_t off = 0;
  auto alloc = [&](size_t n) { float* p = ws + off; off += n; return p; };
  const size_t BIGN = (size_t)B_ * DD * ISEG * DM;          // 50,331,648 floats
  Work w;
  w.S0 = alloc(BIGN); w.S1 = alloc(BIGN); w.S2 = alloc(BIGN); w.S3 = alloc(BIGN);
  w.H  = alloc(2 * BIGN);
  float* e0 = alloc(BIGN);
  float* e1 = alloc(BIGN);
  float* e2 = alloc((size_t)B_ * DD * 12 * DM);
  float* e3 = alloc((size_t)B_ * DD * 3 * DM);
  float* MG = alloc((size_t)B_ * DD * 12 * DM);
  const size_t RN = (size_t)B_ * ISEG * FACTOR_ * DM;
  w.RB = alloc(RN); w.RQ = alloc(RN); w.RA = alloc(RN); w.RO = alloc(RN);
  float* DEC0 = alloc((size_t)B_ * DD * OSEG * DM);
  float* DEC1 = alloc((size_t)B_ * DD * OSEG * DM);
  float* FIN  = alloc((size_t)B_ * DD * OSEG * SL);
  float* XE   = alloc((size_t)B_ * DD * ISEG * SL);

  // ---- DSW embedding ----
  const float* xseq    = (const float*)d_in[373];
  Dense dsw            = getDense(d_in, 237);
  const float* enc_pos = (const float*)d_in[370];
  LNP pre              = getLN(d_in, 371);
  // (b,1152,32) -> (b,32,48,24)
  permute4(st, XE, xseq, (long long)B_ * DD * ISEG * SL, DD, ISEG, SL,
           (long long)1152 * DD, 1, (long long)SL * DD, DD);
  gemm(st, w.S0, XE, dsw.w, dsw.b, B_ * DD * ISEG, DM, SL, 0);
  bcast(st, w.S0, enc_pos, (long long)B_ * DD * ISEG * DM, (long long)DD * ISEG * DM, 1);
  ln512(st, e0, w.S0, nullptr, pre, B_ * DD * ISEG, 1e-3f);   // keras eps

  // ---- Encoder ----
  tsa_fwd(st, e1, e0, B_, DD, ISEG, getTSA(d_in, 239), w);
  {  // block 1: seg-merge 48->12 (pure reshape to (..,12,2048)), then TSA
    Dense lin = getDense(d_in, 280); LNP nm = getLN(d_in, 282);
    ln2048(st, w.S0, e1, nullptr, nm, B_ * DD * 12);
    gemm(st, MG, w.S0, lin.w, lin.b, B_ * DD * 12, DM, 4 * DM, 0);
    tsa_fwd(st, e2, MG, B_, DD, 12, getTSA(d_in, 284), w);
  }
  {  // block 2: seg-merge 12->3
    Dense lin = getDense(d_in, 325); LNP nm = getLN(d_in, 327);
    ln2048(st, w.S0, e2, nullptr, nm, B_ * DD * 3);
    gemm(st, MG, w.S0, lin.w, lin.b, B_ * DD * 3, DM, 4 * DM, 0);
    tsa_fwd(st, e3, MG, B_, DD, 3, getTSA(d_in, 329), w);
  }

  // ---- Decoder ----
  const float* dec_pos = (const float*)d_in[236];
  bcast(st, DEC0, dec_pos, (long long)B_ * DD * OSEG * DM, (long long)DD * OSEG * DM, 0);
  fill0(st, FIN, (long long)B_ * DD * OSEG * SL);
  const float* encs[4] = { e0, e1, e2, e3 };
  const int segE[4]    = { ISEG, ISEG, 12, 3 };
  for (int l = 0; l < 4; ++l) {
    DecP dp = getDec(d_in, l * 59);
    tsa_fwd(st, DEC1, DEC0, B_, DD, OSEG, dp.tsa, w);
    const int G = B_ * DD, Lq = OSEG, Lk = segE[l];
    mha(st, w.S0, DEC1, encs[l], G, Lq, Lk, dp.cross, w.S0, w.S1, w.S2, w.S3);
    ln512(st, w.S2, DEC1, w.S0, dp.n1, G * Lq);                 // xq
    gemm(st, w.H,  w.S2, dp.ml1.w, dp.ml1.b, G * Lq, DM, DM, 1);
    gemm(st, w.S0, w.H,  dp.ml2.w, dp.ml2.b, G * Lq, DM, DM, 0);
    ln512(st, DEC0, w.S2, w.S0, dp.n2, G * Lq);                 // dec_out
    gemm(st, FIN, DEC0, dp.pred.w, dp.pred.b, G * Lq, SL, DM, 2);  // final +=
  }

  // ---- final: (b,32,24,24) -> (b,24,24,32) -> (b,576,32) ----
  permute4(st, (float*)d_out, FIN, (long long)B_ * OSEG * SL * DD, OSEG, SL, DD,
           (long long)DD * OSEG * SL, SL, 1, (long long)OSEG * SL);
}
